// SpanRepLayer_69303592288786
// MI455X (gfx1250) — compile-verified, matrix operand order checked by reading, not run on hardware
//
#include <hip/hip_runtime.h>
#include <hip/hip_bf16.h>
#include <stdint.h>

// ---------------------------------------------------------------------------
// CDNA5 (gfx1250) bf16 WMMA implementation of SpanRepLayer.
//   s_rep = relu( relu(h@W1s+b1s)@W2s+b2s )   (relu folded from cat-relu)
//   e_rep = relu( relu(h@W1e+b1e)@W2e+b2e )
//   out   = relu(gather_cat@Wo1+bo1)@Wo2+bo2  (gather fused into A-loader)
// GEMM core: v_wmma_f32_16x16x32_bf16, f32 accumulate, async global->LDS
// double-buffered staging (ASYNCcnt), weights pre-transposed so both A and B
// fragments load as ds_load_b128.
// ---------------------------------------------------------------------------

typedef __attribute__((ext_vector_type(16))) __bf16 v16bf;
typedef __attribute__((ext_vector_type(8)))  float  v8f;
typedef __attribute__((ext_vector_type(4)))  int    v4i;

// address-space-qualified vector types for the async-LDS builtin
typedef v4i __attribute__((address_space(1))) gv4i;   // global (AS1)
typedef v4i __attribute__((address_space(3))) lv4i;   // LDS    (AS3)

#define TILE_M 128
#define TILE_N 128
#define TILE_K 32
#define LDK    (TILE_K + 8)   // padded row: 40 halves = 80B (16B aligned)
#define THREADS 256           // 8 waves (wave32): 2 (M) x 4 (N) wave grid

__device__ __forceinline__ unsigned short f2bf(float f) {
  unsigned int u = __float_as_uint(f);
  u += 0x7FFFu + ((u >> 16) & 1u);        // round-to-nearest-even
  return (unsigned short)(u >> 16);
}

// ---- CDNA5 async global->LDS copy (16B per lane), guarded fallback --------
__device__ __forceinline__ void async_copy16(const unsigned short* gsrc,
                                             unsigned short* ldst) {
#if __has_builtin(__builtin_amdgcn_global_load_async_to_lds_b128)
  __builtin_amdgcn_global_load_async_to_lds_b128(
      (gv4i*)gsrc, (lv4i*)ldst, 0, 0);
#else
  *(uint4*)ldst = *(const uint4*)gsrc;
#endif
}

__device__ __forceinline__ void wait_async_le4() {
#if __has_builtin(__builtin_amdgcn_s_wait_asynccnt)
  __builtin_amdgcn_s_wait_asynccnt(4);
#elif __has_builtin(__builtin_amdgcn_global_load_async_to_lds_b128)
  asm volatile("s_wait_asynccnt 0x4" ::: "memory");
#endif
}
__device__ __forceinline__ void wait_async_0() {
#if __has_builtin(__builtin_amdgcn_s_wait_asynccnt)
  __builtin_amdgcn_s_wait_asynccnt(0);
#elif __has_builtin(__builtin_amdgcn_global_load_async_to_lds_b128)
  asm volatile("s_wait_asynccnt 0x0" ::: "memory");
#endif
}

// ---- conversions ----------------------------------------------------------
__global__ void cvt_f32_bf16(const float* __restrict__ in,
                             unsigned short* __restrict__ out, int n) {
  int i = blockIdx.x * blockDim.x + threadIdx.x;
  int stride = gridDim.x * blockDim.x;
  for (; i < n; i += stride) out[i] = f2bf(in[i]);
}

// W[K,N] f32 -> Wt[N,K] bf16 (coalesced reads, strided writes; one-time cost)
__global__ void cvt_f32_bf16_T(const float* __restrict__ in,
                               unsigned short* __restrict__ out,
                               int K, int N) {
  int i = blockIdx.x * blockDim.x + threadIdx.x;
  int total = K * N;
  int stride = gridDim.x * blockDim.x;
  for (; i < total; i += stride) {
    int k = i / N, n = i - k * N;
    out[(size_t)n * K + k] = f2bf(in[i]);
  }
}

// 16-bit WMMA operand fragment from an LDS row (ISA 7.12.2 layout):
// element-pair vr -> k = koff + (vr<4 ? 2vr : 2vr+8); lanes<16 koff=0 else 8.
__device__ __forceinline__ v16bf load_frag(const unsigned short* rowbase, int koff) {
  union { v16bf v; unsigned int u[8]; } f;
#pragma unroll
  for (int vr = 0; vr < 8; ++vr) {
    int k = koff + ((vr < 4) ? (2 * vr) : (2 * vr + 8));
    f.u[vr] = *(const unsigned int*)(rowbase + k);
  }
  return f.v;
}

// ---------------------------------------------------------------------------
// C[M,N] = act( A[M,K] @ Wt[N,K]^T + bias[N] )
// GATHER: A row m = concat(start_rep[idx0(m)], end_rep[idx1(m)]) (reps are
//         pre-relu'd), K/2 elements per half, per-batch row tables.
// ---------------------------------------------------------------------------
template<bool GATHER, bool RELU_OUT, bool STORE_BF16>
__global__ __launch_bounds__(THREADS)
void gemm_bf16_wmma(const unsigned short* __restrict__ A,   // [M,K] (or start_rep)
                    const unsigned short* __restrict__ A2,  // end_rep (GATHER)
                    const int* __restrict__ spans,          // [M,2]   (GATHER)
                    const unsigned short* __restrict__ Wt,  // [N,K] bf16 (transposed)
                    const float* __restrict__ bias,         // [N] f32
                    void* __restrict__ Cout,                // bf16 or f32 [M,N]
                    int M, int N, int K,
                    int rows_per_batch, int spans_per_batch)
{
  __shared__ __align__(16) unsigned short lA [2][TILE_M][LDK];  // 2 x 10KB
  __shared__ __align__(16) unsigned short lBt[2][TILE_N][LDK];  // 2 x 10KB

  const int t     = threadIdx.x;
  const int lane  = t & 31;
  const int wave  = t >> 5;
  const int waveM = wave & 1;          // 0..1
  const int waveN = wave >> 1;         // 0..3
  const int lhalf = lane & 15;
  const int koff  = (lane < 16) ? 0 : 8;

  const int mBase = blockIdx.y * TILE_M;
  const int nBase = blockIdx.x * TILE_N;
  const int half  = K >> 1;            // = D for the gather GEMM

  // stage one K-tile (A + Bt) into LDS buffer `buf` via async copies.
  auto stage = [&](int buf, int k0) {
#pragma unroll
    for (int i = 0; i < 2; ++i) {          // A: 512 chunks of 8 halves
      int idx = t + i * THREADS;
      int row = idx >> 2;                  // 4 chunks per 32-half row
      int col = (idx & 3) << 3;
      const unsigned short* gsrc;
      if (!GATHER) {
        gsrc = A + (size_t)(mBase + row) * K + (k0 + col);
      } else {
        int gs = mBase + row;
        int b  = gs / spans_per_batch;
        int kk = k0 + col;
        if (kk < half) {
          int r0 = spans[(size_t)gs * 2 + 0];
          gsrc = A  + (size_t)(b * rows_per_batch + r0) * (size_t)half + kk;
        } else {
          int r1 = spans[(size_t)gs * 2 + 1];
          gsrc = A2 + (size_t)(b * rows_per_batch + r1) * (size_t)half + (kk - half);
        }
      }
      async_copy16(gsrc, &lA[buf][row][col]);
    }
#pragma unroll
    for (int i = 0; i < 2; ++i) {          // Bt: same geometry as A
      int idx = t + i * THREADS;
      int row = idx >> 2;
      int col = (idx & 3) << 3;
      async_copy16(Wt + (size_t)(nBase + row) * K + (k0 + col),
                   &lBt[buf][row][col]);
    }
  };

  v8f acc[4][2];
  const v8f vzero = {0.f,0.f,0.f,0.f,0.f,0.f,0.f,0.f};
#pragma unroll
  for (int i = 0; i < 4; ++i)
#pragma unroll
    for (int j = 0; j < 2; ++j) acc[i][j] = vzero;

  stage(0, 0);                              // prologue
  int buf = 0;
  for (int k0 = 0; k0 < K; k0 += TILE_K, buf ^= 1) {
    if (k0 + TILE_K < K) {
      stage(buf ^ 1, k0 + TILE_K);          // stream next tile...
      wait_async_le4();                     // ...while waiting only for current
    } else {
      wait_async_0();
    }
    __syncthreads();

    v16bf afrag[4];
#pragma unroll
    for (int mt = 0; mt < 4; ++mt)
      afrag[mt] = load_frag(&lA[buf][waveM * 64 + mt * 16 + lhalf][0], koff);
    v16bf bfrag[2];
#pragma unroll
    for (int nt = 0; nt < 2; ++nt)
      bfrag[nt] = load_frag(&lBt[buf][waveN * 32 + nt * 16 + lhalf][0], koff);

#pragma unroll
    for (int mt = 0; mt < 4; ++mt)
#pragma unroll
      for (int nt = 0; nt < 2; ++nt)
        acc[mt][nt] = __builtin_amdgcn_wmma_f32_16x16x32_bf16(
            false, afrag[mt], false, bfrag[nt],
            (short)0, acc[mt][nt], false, false);

    __syncthreads();                        // protect buffer being re-staged
  }

  // ---- epilogue: bias + optional relu, store bf16 or f32 -------------------
#pragma unroll
  for (int mt = 0; mt < 4; ++mt) {
#pragma unroll
    for (int nt = 0; nt < 2; ++nt) {
      int colg = nBase + waveN * 32 + nt * 16 + lhalf;
      float bv = bias[colg];
#pragma unroll
      for (int r = 0; r < 8; ++r) {
        int rowg = mBase + waveM * 64 + mt * 16 + ((lane < 16) ? r : r + 8);
        float v = acc[mt][nt][r] + bv;
        if (RELU_OUT) v = fmaxf(v, 0.0f);
        if (STORE_BF16)
          ((unsigned short*)Cout)[(size_t)rowg * N + colg] = f2bf(v);
        else
          ((float*)Cout)[(size_t)rowg * N + colg] = v;
      }
    }
  }
}

// ---------------------------------------------------------------------------
extern "C" void kernel_launch(void* const* d_in, const int* in_sizes, int n_in,
                              void* d_out, int out_size, void* d_ws, size_t ws_size,
                              hipStream_t stream) {
  (void)in_sizes; (void)n_in; (void)out_size; (void)ws_size;
  const int B = 4, L = 2048, D = 768, S = 16384;
  const int BL = B * L;        // 8192 tokens
  const int BS = B * S;        // 65536 spans
  const int H  = 4 * D;        // 3072

  const float* h        = (const float*)d_in[0];
  const int*   span_idx = (const int*)d_in[1];
  const float* sw1 = (const float*)d_in[2];
  const float* sb1 = (const float*)d_in[3];
  const float* sw2 = (const float*)d_in[4];
  const float* sb2 = (const float*)d_in[5];
  const float* ew1 = (const float*)d_in[6];
  const float* eb1 = (const float*)d_in[7];
  const float* ew2 = (const float*)d_in[8];
  const float* eb2 = (const float*)d_in[9];
  const float* ow1 = (const float*)d_in[10];
  const float* ob1 = (const float*)d_in[11];
  const float* ow2 = (const float*)d_in[12];
  const float* ob2 = (const float*)d_in[13];

  // ---- carve workspace (bf16 buffers, 256B aligned) ------------------------
  char* ws = (char*)d_ws;
  size_t off = 0;
  auto carve = [&](size_t elems) {
    unsigned short* p = (unsigned short*)(ws + off);
    off = (off + elems * 2 + 255) & ~(size_t)255;
    return p;
  };
  unsigned short* h_bf   = carve((size_t)BL * D);
  unsigned short* sw1_t  = carve((size_t)D * H);     // [H,D] layout
  unsigned short* sw2_t  = carve((size_t)H * D);     // [D,H] layout
  unsigned short* ew1_t  = carve((size_t)D * H);
  unsigned short* ew2_t  = carve((size_t)H * D);
  unsigned short* ow1_t  = carve((size_t)2 * D * D); // [D,2D] layout
  unsigned short* ow2_t  = carve((size_t)D * D);     // [D,D]  layout
  unsigned short* hid    = carve((size_t)BL * H);    // reused start & end
  unsigned short* s_rep  = carve((size_t)BL * D);    // stored pre-relu'd
  unsigned short* e_rep  = carve((size_t)BL * D);    // stored pre-relu'd
  unsigned short* hid2   = carve((size_t)BS * D);

  auto cvt = [&](const float* src, unsigned short* dst, int n) {
    int blocks = (n + 255) / 256; if (blocks > 8192) blocks = 8192;
    cvt_f32_bf16<<<blocks, 256, 0, stream>>>(src, dst, n);
  };
  auto cvtT = [&](const float* src, unsigned short* dst, int K, int N) {
    int n = K * N;
    int blocks = (n + 255) / 256; if (blocks > 8192) blocks = 8192;
    cvt_f32_bf16_T<<<blocks, 256, 0, stream>>>(src, dst, K, N);
  };
  cvt (h,   h_bf, BL * D);
  cvtT(sw1, sw1_t, D, H);       // [D,H] -> [H,D]
  cvtT(sw2, sw2_t, H, D);       // [H,D] -> [D,H]
  cvtT(ew1, ew1_t, D, H);
  cvtT(ew2, ew2_t, H, D);
  cvtT(ow1, ow1_t, 2 * D, D);   // [2D,D] -> [D,2D]
  cvtT(ow2, ow2_t, D, D);

  dim3 blk(THREADS);
  // start MLP (note: rep GEMMs store relu(rep) — relu(cat) folded elementwise)
  gemm_bf16_wmma<false, true,  true><<<dim3(H / TILE_N, BL / TILE_M), blk, 0, stream>>>(
      h_bf, nullptr, nullptr, sw1_t, sb1, hid, BL, H, D, L, S);
  gemm_bf16_wmma<false, true,  true><<<dim3(D / TILE_N, BL / TILE_M), blk, 0, stream>>>(
      hid, nullptr, nullptr, sw2_t, sb2, s_rep, BL, D, H, L, S);
  // end MLP (reuses hid)
  gemm_bf16_wmma<false, true,  true><<<dim3(H / TILE_N, BL / TILE_M), blk, 0, stream>>>(
      h_bf, nullptr, nullptr, ew1_t, eb1, hid, BL, H, D, L, S);
  gemm_bf16_wmma<false, true,  true><<<dim3(D / TILE_N, BL / TILE_M), blk, 0, stream>>>(
      hid, nullptr, nullptr, ew2_t, eb2, e_rep, BL, D, H, L, S);
  // out MLP: gather+concat fused into async A-loader (K = 2D = 1536)
  gemm_bf16_wmma<true,  true,  true><<<dim3(D / TILE_N, BS / TILE_M), blk, 0, stream>>>(
      s_rep, e_rep, span_idx, ow1_t, ob1, hid2, BS, D, 2 * D, L, S);
  // final projection -> f32 output
  gemm_bf16_wmma<false, false, false><<<dim3(D / TILE_N, BS / TILE_M), blk, 0, stream>>>(
      hid2, nullptr, nullptr, ow2_t, ob2, d_out, BS, D, D, L, S);
}
